// PredictBlock_82841329205803
// MI455X (gfx1250) — matrix operand order repaired and checked by reading, not compiled, not run on hardware
//
#include <hip/hip_runtime.h>
#include <math.h>
#include <stdint.h>

// Problem constants (match reference)
#define Bb 2
#define Nn 2048
#define Mm 2048
#define Cc 1024
#define Hh 16
#define Dd 32
#define HDm 512
#define FFm 4096

typedef __attribute__((ext_vector_type(16))) __bf16 v16bf;
typedef __attribute__((ext_vector_type(8)))  __bf16 bf16x8;
typedef __attribute__((ext_vector_type(8)))  float  v8f;

__device__ __forceinline__ v16bf cat16(bf16x8 lo, bf16x8 hi) {
  return __builtin_shufflevector(lo, hi, 0,1,2,3,4,5,6,7,8,9,10,11,12,13,14,15);
}
__device__ __forceinline__ v8f wmma_bf16(v16bf a, v16bf b, v8f c) {
  return __builtin_amdgcn_wmma_f32_16x16x32_bf16(false, a, false, b, (short)0, c, false, false);
}

// ---- CDNA5 async Global->LDS copy engine (ASYNCcnt) via inline asm ----
// Generic pointer to LDS = {aperture_hi, lds_byte_offset[31:0]} (ISA 10.2), so the
// low 32 bits of a __shared__ address are the LDS byte address the instruction wants.
__device__ __forceinline__ uint32_t lds_addr32(const void* p) {
  return (uint32_t)(uintptr_t)p;
}
__device__ __forceinline__ void async_ld128(uint32_t lds, const void* g) {
  asm volatile("global_load_async_to_lds_b128 %0, %1, off" :: "v"(lds), "v"(g) : "memory");
}
__device__ __forceinline__ void async_ld64(uint32_t lds, const void* g) {
  asm volatile("global_load_async_to_lds_b64 %0, %1, off" :: "v"(lds), "v"(g) : "memory");
}
__device__ __forceinline__ void wait_async0() {
  asm volatile("s_wait_asynccnt 0x0" ::: "memory");
}

// ---------------- elementwise / repack kernels ----------------

// wq [H, C, D] f32 -> WtT [H*D, C] bf16  (transposed for GEMM B staging)
__global__ void repack_qkv_wT_kernel(const float* __restrict__ in, __bf16* __restrict__ out) {
  int i = blockIdx.x * 256 + threadIdx.x;
  int total = Hh * Cc * Dd;
  if (i >= total) return;
  int d = i & 31;
  int c = (i >> 5) % Cc;
  int h = i / (Cc * Dd);
  out[(size_t)(h * Dd + d) * Cc + c] = (__bf16)in[i];
}

// in [K, N] f32 -> out [N, K] bf16 (transpose + convert)
__global__ void cvtT_kernel(const float* __restrict__ in, __bf16* __restrict__ out, int K, int N) {
  int i = blockIdx.x * 256 + threadIdx.x;
  if (i >= K * N) return;
  int k = i / N, n = i % N;
  out[(size_t)n * K + k] = (__bf16)in[i];
}

// vh [B*M, H*D] bf16 -> vT [B,H,D,M] bf16
__global__ void repack_vT_kernel(const __bf16* __restrict__ vh, __bf16* __restrict__ vT, int Mk) {
  int i = blockIdx.x * 256 + threadIdx.x;
  int total = Bb * Mk * HDm;
  if (i >= total) return;
  int d  = i & 31;
  int hh = (i >> 5) & 15;
  int m  = (i >> 9) % Mk;
  int bq = i / (Mk * HDm);
  vT[(((size_t)bq * Hh + hh) * Dd + d) * Mk + m] = vh[i];
}

__global__ void add2_kernel(const float* __restrict__ a, const float* __restrict__ b,
                            float* __restrict__ out, int n) {
  int i = blockIdx.x * 256 + threadIdx.x;
  if (i < n) out[i] = a[i] + b[i];
}

// ---------------- LayerNorm: one block per row of 1024 ----------------

__global__ void layernorm_kernel(const float* __restrict__ in, const float* __restrict__ g,
                                 const float* __restrict__ bt, float* __restrict__ outF,
                                 __bf16* __restrict__ outB) {
  __shared__ float s1[256], s2[256];
  const int row = blockIdx.x;
  const int tid = threadIdx.x;
  const float* x = in + (size_t)row * Cc;
  float v[4];
  float sum = 0.f, sq = 0.f;
#pragma unroll
  for (int i = 0; i < 4; ++i) {
    v[i] = x[tid + i * 256];
    sum += v[i];
    sq  += v[i] * v[i];
  }
  s1[tid] = sum; s2[tid] = sq;
  __syncthreads();
  for (int s = 128; s > 0; s >>= 1) {
    if (tid < s) { s1[tid] += s1[tid + s]; s2[tid] += s2[tid + s]; }
    __syncthreads();
  }
  float mean = s1[0] * (1.f / Cc);
  float var  = s2[0] * (1.f / Cc) - mean * mean;
  float rstd = rsqrtf(var + 1e-5f);
#pragma unroll
  for (int i = 0; i < 4; ++i) {
    int c = tid + i * 256;
    float o = (v[i] - mean) * rstd * g[c] + bt[c];
    if (outF) outF[(size_t)row * Cc + c] = o;
    if (outB) outB[(size_t)row * Cc + c] = (__bf16)o;
  }
}

// ---------------- WMMA GEMM ----------------
// out = act(A[Mr,K] * W[K,Nc] + bias) + resid, with W supplied TRANSPOSED as Wt[Nc,K].
// 128x128 block tile, 8 waves each 32x64 (2 A-frags x 4 B-frags = 8 WMMAs/k-step),
// double-buffered LDS staged by the async Global->LDS engine, one barrier per k-step.

__global__ void gemm_bf16_kernel(const __bf16* __restrict__ A, const __bf16* __restrict__ Wt,
                                 const float* __restrict__ bias, const float* __restrict__ resid,
                                 float* __restrict__ outF, __bf16* __restrict__ outB,
                                 int Mr, int K, int Nc, int act) {
  __shared__ __bf16 As[2][128][40];   // [buf][row][k], padded rows (80B stride)
  __shared__ __bf16 Bs[2][128][40];   // [buf][n][k]
  const int tid  = threadIdx.x;
  const int wave = tid >> 5;
  const int lane = tid & 31;
  const int hlf  = lane >> 4;
  const int l16  = lane & 15;
  const int wrow = (wave & 3) * 32;   // 0,32,64,96
  const int wcol = (wave >> 2) * 64;  // 0,64
  const int m0 = blockIdx.y * 128;
  const int n0 = blockIdx.x * 128;

  // staging assignment: each thread owns one row-half (16 halfs = 2 x b128)
  const int sr = tid >> 1;            // 0..127
  const int sc = (tid & 1) * 16;      // 0 or 16
  const __bf16* gA = A  + (size_t)(m0 + sr) * K + sc;
  const __bf16* gB = Wt + (size_t)(n0 + sr) * K + sc;

  v8f acc[2][4];
#pragma unroll
  for (int ai = 0; ai < 2; ++ai)
#pragma unroll
    for (int f = 0; f < 4; ++f) acc[ai][f] = (v8f){};

  // prologue: async-stage k-step 0 into buffer 0
  async_ld128(lds_addr32(&As[0][sr][sc]),     gA);
  async_ld128(lds_addr32(&As[0][sr][sc + 8]), gA + 8);
  async_ld128(lds_addr32(&Bs[0][sr][sc]),     gB);
  async_ld128(lds_addr32(&Bs[0][sr][sc + 8]), gB + 8);

  int cur = 0;
  for (int kk = 0; kk < K; kk += 32) {
    wait_async0();        // own async writes to LDS complete
    __syncthreads();      // all waves' staging of buf[cur] visible
    // async-stage next k-step into the other buffer (overlaps WMMA below)
    if (kk + 32 < K) {
      int nxt = cur ^ 1;
      int ko = kk + 32;
      async_ld128(lds_addr32(&As[nxt][sr][sc]),     gA + ko);
      async_ld128(lds_addr32(&As[nxt][sr][sc + 8]), gA + ko + 8);
      async_ld128(lds_addr32(&Bs[nxt][sr][sc]),     gB + ko);
      async_ld128(lds_addr32(&Bs[nxt][sr][sc + 8]), gB + ko + 8);
    }

    v16bf a0 = cat16(*(const bf16x8*)&As[cur][wrow + l16][8 * hlf],
                     *(const bf16x8*)&As[cur][wrow + l16][8 * hlf + 16]);
    v16bf a1 = cat16(*(const bf16x8*)&As[cur][wrow + 16 + l16][8 * hlf],
                     *(const bf16x8*)&As[cur][wrow + 16 + l16][8 * hlf + 16]);
#pragma unroll
    for (int f = 0; f < 4; ++f) {
      int n = wcol + f * 16 + l16;
      v16bf bf = cat16(*(const bf16x8*)&Bs[cur][n][8 * hlf],
                       *(const bf16x8*)&Bs[cur][n][8 * hlf + 16]);
      acc[0][f] = wmma_bf16(a0, bf, acc[0][f]);
      acc[1][f] = wmma_bf16(a1, bf, acc[1][f]);
    }
    cur ^= 1;
  }

#pragma unroll
  for (int ai = 0; ai < 2; ++ai) {
#pragma unroll
    for (int r = 0; r < 8; ++r) {
      int grow = m0 + wrow + ai * 16 + r + 8 * hlf;
#pragma unroll
      for (int f = 0; f < 4; ++f) {
        int gcol = n0 + wcol + f * 16 + l16;
        float v = acc[ai][f][r];
        if (bias) v += bias[gcol];
        if (act == 1) v = 0.5f * v * (1.0f + erff(v * 0.70710678118654752f));
        size_t o = (size_t)grow * Nc + gcol;
        if (resid) v += resid[o];
        if (outF) outF[o] = v;
        if (outB) outB[o] = (__bf16)v;
      }
    }
  }
}

// ---------------- Flash-style attention ----------------
// Q: [B,N,H,D] bf16; K: [B,M,H,D] bf16; Vt: [B,H,D,M] bf16; Ob: [B,N,H,D] bf16
// Softmax over full row (unmasked denominator); post-softmax causal mask zeroes j >= i.
// K/V chunks double-buffered via async Global->LDS copies.

__global__ void attention_kernel(const __bf16* __restrict__ Q, const __bf16* __restrict__ Kb,
                                 const __bf16* __restrict__ Vt, __bf16* __restrict__ Ob,
                                 int Nq, int Mk, int causal) {
  __shared__ __bf16 Ks[2][32][40];     // [buf][key][dim]
  __shared__ __bf16 Vs[2][32][40];     // [buf][dim][key]
  __shared__ __bf16 Ps[8][16][40];     // per-wave P tile [row][key]
  const int tid  = threadIdx.x;
  const int wave = tid >> 5;
  const int lane = tid & 31;
  const int hlf  = lane >> 4;
  const int l16  = lane & 15;
  const int h = blockIdx.y;
  const int b = blockIdx.z;
  const int q0 = blockIdx.x * 128 + wave * 16;
  const float scale = 0.17677669529663687f; // 1/sqrt(32)

  // per-thread staging slot: 4 halfs (8B) of K and of V
  const int sIdx = tid * 4;
  const int sKey = sIdx >> 5;      // 0..31
  const int sD0  = sIdx & 31;      // 0,4,..,28
  const __bf16* gK = Kb + ((size_t)b * Mk + sKey) * HDm + h * Dd + sD0;
  const __bf16* gV = Vt + (((size_t)b * Hh + h) * Dd + sKey) * Mk + sD0;

  // Q A-fragment (persistent)
  const size_t qrow = ((size_t)b * Nq + q0 + l16) * HDm + h * Dd;
  bf16x8 qlo = *(const bf16x8*)(Q + qrow + 8 * hlf);
  bf16x8 qhi = *(const bf16x8*)(Q + qrow + 8 * hlf + 16);
  v16bf qa = cat16(qlo, qhi);

  float Mrun[8], L[8];
  v8f o0 = {}; v8f o1 = {};
#pragma unroll
  for (int r = 0; r < 8; ++r) { Mrun[r] = -1e30f; L[r] = 0.f; }

  // prologue: async-stage chunk 0 into buffer 0
  async_ld64(lds_addr32(&Ks[0][sKey][sD0]), gK);
  async_ld64(lds_addr32(&Vs[0][sKey][sD0]), gV);

  int cur = 0;
  for (int cm = 0; cm < Mk; cm += 32) {
    wait_async0();
    __syncthreads();
    if (cm + 32 < Mk) {
      int nxt = cur ^ 1;
      async_ld64(lds_addr32(&Ks[nxt][sKey][sD0]), gK + (size_t)(cm + 32) * HDm);
      async_ld64(lds_addr32(&Vs[nxt][sKey][sD0]), gV + (cm + 32));
    }

    // score B-frags: B[kdim][key]
    v16bf bk0 = cat16(*(const bf16x8*)&Ks[cur][l16][8 * hlf],
                      *(const bf16x8*)&Ks[cur][l16][8 * hlf + 16]);
    v16bf bk1 = cat16(*(const bf16x8*)&Ks[cur][16 + l16][8 * hlf],
                      *(const bf16x8*)&Ks[cur][16 + l16][8 * hlf + 16]);
    // value B-frags: B[key][dim] from Vs[dim][key]
    v16bf bv0 = cat16(*(const bf16x8*)&Vs[cur][l16][8 * hlf],
                      *(const bf16x8*)&Vs[cur][l16][8 * hlf + 16]);
    v16bf bv1 = cat16(*(const bf16x8*)&Vs[cur][16 + l16][8 * hlf],
                      *(const bf16x8*)&Vs[cur][16 + l16][8 * hlf + 16]);

    v8f z = {};
    v8f s0 = wmma_bf16(qa, bk0, z);
    v8f s1 = wmma_bf16(qa, bk1, z);

#pragma unroll
    for (int r = 0; r < 8; ++r) {
      float t0 = s0[r] * scale, t1 = s1[r] * scale;
      float rm = fmaxf(t0, t1);
#pragma unroll
      for (int mm = 1; mm < 16; mm <<= 1) rm = fmaxf(rm, __shfl_xor(rm, mm, 32));
      float mnew = fmaxf(Mrun[r], rm);
      float alpha = expf(Mrun[r] - mnew);
      float p0 = expf(t0 - mnew), p1 = expf(t1 - mnew);
      float rs = p0 + p1;
#pragma unroll
      for (int mm = 1; mm < 16; mm <<= 1) rs += __shfl_xor(rs, mm, 32);
      L[r] = L[r] * alpha + rs;
      Mrun[r] = mnew;
      o0[r] *= alpha; o1[r] *= alpha;
      // post-softmax causal mask (zero where key >= query, incl diagonal)
      int qi = q0 + r + 8 * hlf;
      float pm0 = p0, pm1 = p1;
      if (causal) {
        if (cm + l16 >= qi)      pm0 = 0.f;
        if (cm + 16 + l16 >= qi) pm1 = 0.f;
      }
      Ps[wave][r + 8 * hlf][l16]      = (__bf16)pm0;
      Ps[wave][r + 8 * hlf][16 + l16] = (__bf16)pm1;
    }
    __syncthreads();

    // P A-fragment from own wave's LDS tile
    v16bf pa = cat16(*(const bf16x8*)&Ps[wave][l16][8 * hlf],
                     *(const bf16x8*)&Ps[wave][l16][8 * hlf + 16]);
    o0 = wmma_bf16(pa, bv0, o0);
    o1 = wmma_bf16(pa, bv1, o1);
    cur ^= 1;
  }

#pragma unroll
  for (int r = 0; r < 8; ++r) {
    int qi = q0 + r + 8 * hlf;
    float inv = 1.0f / L[r];
    size_t base = ((size_t)b * Nq + qi) * HDm + h * Dd;
    Ob[base + l16]      = (__bf16)(o0[r] * inv);
    Ob[base + 16 + l16] = (__bf16)(o1[r] * inv);
  }
}

// ---------------- host driver ----------------

extern "C" void kernel_launch(void* const* d_in, const int* in_sizes, int n_in,
                              void* d_out, int out_size, void* d_ws, size_t ws_size,
                              hipStream_t stream) {
  (void)in_sizes; (void)n_in; (void)out_size; (void)ws_size;
  const float* xx   = (const float*)d_in[0];
  const float* ee   = (const float*)d_in[1];
  const float* lnAg = (const float*)d_in[2];
  const float* lnAb = (const float*)d_in[3];
  const float* lnBg = (const float*)d_in[4];
  const float* lnBb = (const float*)d_in[5];

  char* base = (char*)d_ws;
  size_t off = 0;
  auto arena = [&](size_t bytes) -> void* {
    void* p = base + off;
    off = (off + bytes + 255) & ~(size_t)255;
    return p;
  };

  __bf16* wq_b = (__bf16*)arena((size_t)Cc * HDm * 2);   // [HD, C]
  __bf16* wk_b = (__bf16*)arena((size_t)Cc * HDm * 2);
  __bf16* wv_b = (__bf16*)arena((size_t)Cc * HDm * 2);
  __bf16* wo_b = (__bf16*)arena((size_t)HDm * Cc * 2);   // [C, HD]
  __bf16* w2_b = (__bf16*)arena((size_t)Cc * FFm * 2);   // [FF, C]
  __bf16* w3_b = (__bf16*)arena((size_t)FFm * Cc * 2);   // [C, FF]
  float*  xf    = (float*)arena((size_t)Bb * Nn * Cc * 4);
  float*  xf2   = (float*)arena((size_t)Bb * Nn * Cc * 4);
  float*  lnq_f = (float*)arena((size_t)Bb * Nn * Cc * 4);
  __bf16* lnq_b = (__bf16*)arena((size_t)Bb * Nn * Cc * 2);
  __bf16* lny_b = (__bf16*)arena((size_t)Bb * Mm * Cc * 2);
  __bf16* qh    = (__bf16*)arena((size_t)Bb * Nn * HDm * 2);
  __bf16* kh    = (__bf16*)arena((size_t)Bb * Mm * HDm * 2);
  __bf16* vh    = (__bf16*)arena((size_t)Bb * Mm * HDm * 2);
  __bf16* vT    = (__bf16*)arena((size_t)Bb * Mm * HDm * 2);
  __bf16* ob    = (__bf16*)arena((size_t)Bb * Nn * HDm * 2);
  float*  outf  = (float*)arena((size_t)Bb * Nn * Cc * 4);
  __bf16* hb    = (__bf16*)arena((size_t)Bb * Nn * Cc * 2);
  __bf16* hid   = (__bf16*)arena((size_t)Bb * Nn * FFm * 2);
  float*  ccb   = (float*)arena((size_t)Bb * Nn * Cc * 4);

  auto cvtT = [&](const float* in, __bf16* o, int K, int N) {
    int n = K * N;
    cvtT_kernel<<<(n + 255) / 256, 256, 0, stream>>>(in, o, K, N);
  };
  auto ln = [&](const float* in, const float* g, const float* bt, float* oF, __bf16* oB, int rows) {
    layernorm_kernel<<<rows, 256, 0, stream>>>(in, g, bt, oF, oB);
  };
  auto gemm = [&](const __bf16* A, const __bf16* Wt, const float* bias, const float* resid,
                  float* oF, __bf16* oB, int Mr, int K, int Nc, int act) {
    dim3 g(Nc / 128, Mr / 128);
    gemm_bf16_kernel<<<g, 256, 0, stream>>>(A, Wt, bias, resid, oF, oB, Mr, K, Nc, act);
  };

  // one transformer block; pi = index of <p>_ln1_g; xres = query-side f32 input; ysrc = KV-side f32
  auto run_block = [&](int pi, const float* xres, const float* ysrc, int Mk, int causal,
                       float* cc_out) {
    const float* ln1g = (const float*)d_in[pi + 0];
    const float* ln1b = (const float*)d_in[pi + 1];
    const float* ln2g = (const float*)d_in[pi + 2];
    const float* ln2b = (const float*)d_in[pi + 3];
    const float* ln3g = (const float*)d_in[pi + 4];
    const float* ln3b = (const float*)d_in[pi + 5];
    const float* wq = (const float*)d_in[pi + 6];
    const float* wk = (const float*)d_in[pi + 7];
    const float* wv = (const float*)d_in[pi + 8];
    const float* wo = (const float*)d_in[pi + 9];
    const float* bo = (const float*)d_in[pi + 10];
    const float* w2 = (const float*)d_in[pi + 11];
    const float* b2 = (const float*)d_in[pi + 12];
    const float* w3 = (const float*)d_in[pi + 13];
    const float* b3 = (const float*)d_in[pi + 14];

    int nqkv = Hh * Cc * Dd;
    repack_qkv_wT_kernel<<<(nqkv + 255) / 256, 256, 0, stream>>>(wq, wq_b);
    repack_qkv_wT_kernel<<<(nqkv + 255) / 256, 256, 0, stream>>>(wk, wk_b);
    repack_qkv_wT_kernel<<<(nqkv + 255) / 256, 256, 0, stream>>>(wv, wv_b);
    cvtT(wo, wo_b, HDm, Cc);   // [512,1024] -> [1024,512]
    cvtT(w2, w2_b, Cc, FFm);   // [1024,4096] -> [4096,1024]
    cvtT(w3, w3_b, FFm, Cc);   // [4096,1024] -> [1024,4096]

    ln(xres, ln1g, ln1b, lnq_f, lnq_b, Bb * Nn);
    ln(ysrc, ln2g, ln2b, nullptr, lny_b, Bb * Mk);

    gemm(lnq_b, wq_b, nullptr, nullptr, nullptr, qh, Bb * Nn, Cc, HDm, 0);
    gemm(lny_b, wk_b, nullptr, nullptr, nullptr, kh, Bb * Mk, Cc, HDm, 0);
    gemm(lny_b, wv_b, nullptr, nullptr, nullptr, vh, Bb * Mk, Cc, HDm, 0);
    {
      int n = Bb * Mk * HDm;
      repack_vT_kernel<<<(n + 255) / 256, 256, 0, stream>>>(vh, vT, Mk);
    }
    attention_kernel<<<dim3(Nn / 128, Hh, Bb), 256, 0, stream>>>(qh, kh, vT, ob, Nn, Mk, causal);

    // out = ln1(x) + o @ wo + bo
    gemm(ob, wo_b, bo, lnq_f, outf, nullptr, Bb * Nn, HDm, Cc, 0);
    // h = gelu(ln3(out) @ w2 + b2)
    ln(outf, ln3g, ln3b, nullptr, hb, Bb * Nn);
    gemm(hb, w2_b, b2, nullptr, nullptr, hid, Bb * Nn, Cc, FFm, 1);
    // cc = out + h @ w3 + b3
    gemm(hid, w3_b, b3, outf, cc_out, nullptr, Bb * Nn, FFm, Cc, 0);
  };

  const int nx = Bb * Nn * Cc;

  // x = ln(xx, lnA)
  ln(xx, lnAg, lnAb, xf, nullptr, Bb * Nn);
  // block a1: self-attention, causal
  run_block(6, xf, xf, Mm, 1, ccb);
  // x = x + cc
  add2_kernel<<<(nx + 255) / 256, 256, 0, stream>>>(xf, ccb, xf, nx);
  // x = ln(x, lnB)
  ln(xf, lnBg, lnBb, xf2, nullptr, Bb * Nn);
  // block a2: cross-attention with ee, no mask
  run_block(21, xf2, ee, Mm, 0, ccb);
  // final x = x + cc  -> d_out[0 : B*N*C]
  add2_kernel<<<(nx + 255) / 256, 256, 0, stream>>>(xf2, ccb, (float*)d_out, nx);
  // second output: ee passthrough
  hipMemcpyAsync((float*)d_out + (size_t)nx, ee, (size_t)Bb * Mm * Cc * sizeof(float),
                 hipMemcpyDeviceToDevice, stream);
}
